// AngularMarginLoss_22728966930439
// MI455X (gfx1250) — matrix operand order repaired and checked by reading, not compiled, not run on hardware
//
#include <hip/hip_runtime.h>
#include <hip/hip_bf16.h>
#include <math.h>

typedef __attribute__((ext_vector_type(16))) _Float16 v16h;
typedef __attribute__((ext_vector_type(8)))  _Float16 v8h;
typedef __attribute__((ext_vector_type(4)))  _Float16 v4h;
typedef __attribute__((ext_vector_type(8)))  float    v8f;
typedef __attribute__((ext_vector_type(4)))  float    v4f;

#define N_ROWS   2048
#define DIM      128
#define N_COLS   100000
#define S_SCALE  64.0f
#define EPS_CLIP 1e-7f
// cos(0.5), sin(0.5)
#define COS_M 0.8775825618903728f
#define SIN_M 0.4794255386042030f

// GEMM tiling: 125 column-chunk blocks, each handling 25 iterations of 32 cols
#define COL_BLKS   125
#define GEMM_ITERS 25            // (N_COLS/32) / COL_BLKS = 3125/125, exact

// Workspace layout (bytes):
//   [0      .. 8191 ]  row_sums float[2048]
//   [8192   .. 16383]  tgt      float[2048]
//   [16384  .. +512K]  x_h      f16[2048*128]
//   [540672 .. +25.6M] w_h      f16[100000*128]
#define WS_TGT_OFF  8192
#define WS_XH_OFF   16384
#define WS_WH_OFF   540672

// ---------------------------------------------------------------------------
// Kernel 1: L2-normalize each row of x, convert to f16; zero accumulators.
// ---------------------------------------------------------------------------
__global__ void aml_normalize_x(const float* __restrict__ x,
                                float* __restrict__ row_sums,
                                _Float16* __restrict__ xh) {
    const int row = blockIdx.x;
    const int tid = threadIdx.x;           // 0..127
    float v  = x[row * DIM + tid];
    float ss = v * v;
    #pragma unroll
    for (int m = 16; m >= 1; m >>= 1) ss += __shfl_xor(ss, m, 32);
    __shared__ float part[4];
    if ((tid & 31) == 0) part[tid >> 5] = ss;
    __syncthreads();
    const float tot = part[0] + part[1] + part[2] + part[3];
    const float r   = rsqrtf(tot);
    xh[row * DIM + tid] = (_Float16)(v * r);
    if (tid == 0) row_sums[row] = 0.0f;
}

// ---------------------------------------------------------------------------
// Kernel 2: convert W (fp32 [C, D] row-major) to f16.
// ---------------------------------------------------------------------------
__global__ void aml_convert_w(const float* __restrict__ W,
                              _Float16* __restrict__ wh, int total4) {
    int i = blockIdx.x * blockDim.x + threadIdx.x;
    const int stride = gridDim.x * blockDim.x;
    for (; i < total4; i += stride) {
        v4f f = ((const v4f*)W)[i];
        v4h h;
        h.x = (_Float16)f.x; h.y = (_Float16)f.y;
        h.z = (_Float16)f.z; h.w = (_Float16)f.w;
        ((v4h*)wh)[i] = h;
    }
}

// ---------------------------------------------------------------------------
// Kernel 3: tgt[i] = dot(x_n[i], W[labels[i]]) from the same f16 operands the
// WMMA consumes (f32 accumulate). One wave per row, 4 elements per lane.
// ---------------------------------------------------------------------------
__global__ void aml_target(const _Float16* __restrict__ xh,
                           const _Float16* __restrict__ wh,
                           const int* __restrict__ labels,
                           float* __restrict__ tgt) {
    const int row  = blockIdx.x * 4 + (threadIdx.x >> 5);
    const int lane = threadIdx.x & 31;
    const int lab  = labels[row];
    v4h xv = *(const v4h*)(xh + (size_t)row * DIM + lane * 4);
    v4h wv = *(const v4h*)(wh + (size_t)lab * DIM + lane * 4);
    float s = (float)xv.x * (float)wv.x + (float)xv.y * (float)wv.y
            + (float)xv.z * (float)wv.z + (float)xv.w * (float)wv.w;
    #pragma unroll
    for (int m = 16; m >= 1; m >>= 1) s += __shfl_xor(s, m, 32);
    if (lane == 0) tgt[row] = s;
}

// ---------------------------------------------------------------------------
// GEMM helpers
// ---------------------------------------------------------------------------
__device__ __forceinline__ void load_btile(const _Float16* __restrict__ p0,
                                           v16h* b0, v16h* b1) {
    const _Float16* p1 = p0 + 16 * DIM;
    #pragma unroll
    for (int k = 0; k < 4; ++k) {
        b0[k] = *(const v16h*)(p0 + k * 32);
        b1[k] = *(const v16h*)(p1 + k * 32);
    }
}

__device__ __forceinline__ void gemm_tile(const v16h* a,
                                          const v16h* b0, const v16h* b1,
                                          v8f& acc) {
    v8f c0 = {0.f,0.f,0.f,0.f,0.f,0.f,0.f,0.f};
    v8f c1 = {0.f,0.f,0.f,0.f,0.f,0.f,0.f,0.f};
    #pragma unroll
    for (int k = 0; k < 4; ++k) {
        c0 = __builtin_amdgcn_wmma_f32_16x16x32_f16(
                false, a[k], false, b0[k], (short)0, c0, false, false);
        c1 = __builtin_amdgcn_wmma_f32_16x16x32_f16(
                false, a[k], false, b1[k], (short)0, c1, false, false);
    }
    #pragma unroll
    for (int v = 0; v < 8; ++v)
        acc[v] += __expf(S_SCALE * c0[v]) + __expf(S_SCALE * c1[v]);
}

// ---------------------------------------------------------------------------
// Kernel 4: fused GEMM + exp + row-sum. Block = 256 threads = 8 waves; wave w
// owns rows [blockIdx.y*128 + w*16, +16). Each iteration covers 32 columns as
// two independent WMMA chains; B tiles are double-buffered across iterations.
// sched_barrier(0) fences keep the scheduler from triple-buffering (which
// previously blew past the VGPR budget and spilled the A panel to scratch);
// launch_bounds(256,1) raises the per-wave VGPR cap for the ~190-reg state.
//
// A layout (16-bit A 16x32): lane<16 -> row=lane, K k0..k0+15;
//   lane>=16 -> row=lane-16, K k0+16..k0+31.
// B layout (dense 32x16): lane -> col=lane%16; lanes 0-15 K k0..k0+15,
//   lanes 16-31 K k0+16..k0+31 (16 contiguous halfs per lane).
// C/D: VGPR v, lane l -> (M = v + 8*(l>=16), N = l%16).
// ---------------------------------------------------------------------------
__global__ void __launch_bounds__(256, 1)
aml_gemm_exp(const _Float16* __restrict__ xh,
             const _Float16* __restrict__ wh,
             float* __restrict__ row_sums) {
    const int wave    = threadIdx.x >> 5;       // 0..7
    const int lane    = threadIdx.x & 31;
    const int lhalf   = lane & 15;
    const int hi      = lane >> 4;
    const int rowBase = blockIdx.y * 128 + wave * 16;

    // full A panel (16 rows x 128 K) in registers: 4 x v16h = 32 VGPRs
    v16h a[4];
    {
        const _Float16* arow = xh + (size_t)(rowBase + lhalf) * DIM;
        #pragma unroll
        for (int k = 0; k < 4; ++k) {
            const int k0 = k * 32;
            v8h lo  = *(const v8h*)(arow + k0 + hi * 16);
            v8h hih = *(const v8h*)(arow + k0 + hi * 16 + 8);
            a[k] = __builtin_shufflevector(lo, hih,
                    0,1,2,3,4,5,6,7,8,9,10,11,12,13,14,15);
        }
    }

    v8f acc = {0.f,0.f,0.f,0.f,0.f,0.f,0.f,0.f};

    const size_t step = (size_t)COL_BLKS * 32 * DIM;     // halfs per iteration
    const _Float16* p = wh + (size_t)(blockIdx.x * 32 + lhalf) * DIM + hi * 16;

    v16h b0[4], b1[4], n0[4], n1[4];
    load_btile(p, b0, b1);                               // iter 0

    // steady state: (GEMM_ITERS-1)/2 = 12 double-iterations, then epilogue
    #pragma unroll 1
    for (int i = 0; i < (GEMM_ITERS - 1) / 2; ++i) {
        __builtin_prefetch(p + 2 * step, 0, 1);
        load_btile(p + step, n0, n1);    // issue iter 2i+1 loads
        __builtin_amdgcn_sched_barrier(0);
        gemm_tile(a, b0, b1, acc);       // compute iter 2i (frees b)
        __builtin_amdgcn_sched_barrier(0);
        p += 2 * step;
        load_btile(p, b0, b1);           // issue iter 2i+2 loads into b
        __builtin_amdgcn_sched_barrier(0);
        gemm_tile(a, n0, n1, acc);       // compute iter 2i+1 (frees n)
        __builtin_amdgcn_sched_barrier(0);
    }
    gemm_tile(a, b0, b1, acc);           // iter 24

    // reduce partials across the 16 lanes sharing each row; one atomic per row
    #pragma unroll
    for (int m = 1; m <= 8; m <<= 1) {
        #pragma unroll
        for (int v = 0; v < 8; ++v) acc[v] += __shfl_xor(acc[v], m, 32);
    }
    if (lhalf == 0) {
        #pragma unroll
        for (int v = 0; v < 8; ++v)
            atomicAdd(&row_sums[rowBase + hi * 8 + v], acc[v]);
    }
}

// ---------------------------------------------------------------------------
// Kernel 5: per-row margin logit + log-sum assembly, mean-reduce.
// ---------------------------------------------------------------------------
__global__ void aml_finalize(const float* __restrict__ row_sums,
                             const float* __restrict__ tgt,
                             float* __restrict__ out) {
    const int tid = threadIdx.x;   // 256
    float lsum = 0.0f;
    for (int i = tid; i < N_ROWS; i += 256) {
        const float t  = tgt[i];
        float tc = fminf(fmaxf(t, -1.0f + EPS_CLIP), 1.0f - EPS_CLIP);
        const float num  = S_SCALE * (tc * COS_M -
                            sqrtf(fmaxf(1.0f - tc * tc, 0.0f)) * SIN_M);
        const float excl = row_sums[i] - __expf(S_SCALE * t);
        const float den  = __expf(num) + excl;
        lsum += num - __logf(den);
    }
    __shared__ float sm[256];
    sm[tid] = lsum;
    __syncthreads();
    #pragma unroll
    for (int s = 128; s >= 1; s >>= 1) {
        if (tid < s) sm[tid] += sm[tid + s];
        __syncthreads();
    }
    if (tid == 0) out[0] = -sm[0] / (float)N_ROWS;
}

// ---------------------------------------------------------------------------
extern "C" void kernel_launch(void* const* d_in, const int* in_sizes, int n_in,
                              void* d_out, int out_size, void* d_ws, size_t ws_size,
                              hipStream_t stream) {
    const float* x      = (const float*)d_in[0];
    const float* W      = (const float*)d_in[1];
    const int*   labels = (const int*)d_in[2];
    float*       out    = (float*)d_out;

    char* ws = (char*)d_ws;
    float*    row_sums = (float*)ws;
    float*    tgt      = (float*)(ws + WS_TGT_OFF);
    _Float16* xh       = (_Float16*)(ws + WS_XH_OFF);
    _Float16* wh       = (_Float16*)(ws + WS_WH_OFF);

    aml_normalize_x<<<N_ROWS, DIM, 0, stream>>>(x, row_sums, xh);
    aml_convert_w<<<2048, 256, 0, stream>>>(W, wh, (N_COLS * DIM) / 4);
    aml_target<<<N_ROWS / 4, 128, 0, stream>>>(xh, wh, labels, tgt);
    dim3 grid(COL_BLKS, N_ROWS / 128);   // 125 col-chunks x 16 row-blocks
    aml_gemm_exp<<<grid, 256, 0, stream>>>(xh, wh, row_sums);
    aml_finalize<<<1, 256, 0, stream>>>(row_sums, tgt, out);
}
